// CrossLayerAttention_67430986547826
// MI455X (gfx1250) — matrix operand order repaired and checked by reading, not compiled, not run on hardware
//
#include <hip/hip_runtime.h>
#include <stdint.h>

// Problem constants (B, C_IN, C_OUT, H, W = 4, 64, 64, 64, 64)
#define BATCH 4
#define CIN   64
#define DH    128      // projected dim D = 2*C_OUT
#define HW    4096     // H*W
#define NTOK  4096     // N tokens
#define KTILE 64       // keys per LDS tile
#define WAVES 8        // waves per flash block

typedef __attribute__((ext_vector_type(16))) __bf16          v16bf;
typedef __attribute__((ext_vector_type(8)))  float           v8f;
typedef __attribute__((ext_vector_type(2)))  float           v2f;
typedef __attribute__((ext_vector_type(8)))  unsigned short  us8;

union BF16x16 { v16bf v; us8 h[2]; };

__device__ __forceinline__ unsigned short f2bf(float x) {
    unsigned int u = __float_as_uint(x);
    u = (u + 0x7FFFu + ((u >> 16) & 1u)) >> 16;   // RNE
    return (unsigned short)u;
}

// ---- DPP16 xor-tree reductions over the 16-lane half (stays within DPP row)
__device__ __forceinline__ float red_max16(float x) {
    float t;
    t = __int_as_float(__builtin_amdgcn_update_dpp(0, __float_as_int(x), 0xB1, 0xF, 0xF, true)); // quad_perm(1,0,3,2)
    x = fmaxf(x, t);
    t = __int_as_float(__builtin_amdgcn_update_dpp(0, __float_as_int(x), 0x4E, 0xF, 0xF, true)); // quad_perm(2,3,0,1)
    x = fmaxf(x, t);
    t = __int_as_float(__builtin_amdgcn_update_dpp(0, __float_as_int(x), 0x141, 0xF, 0xF, true)); // row_half_mirror
    x = fmaxf(x, t);
    t = __int_as_float(__builtin_amdgcn_update_dpp(0, __float_as_int(x), 0x140, 0xF, 0xF, true)); // row_mirror
    return fmaxf(x, t);
}
__device__ __forceinline__ float red_sum16(float x) {
    float t;
    t = __int_as_float(__builtin_amdgcn_update_dpp(0, __float_as_int(x), 0xB1, 0xF, 0xF, true));
    x += t;
    t = __int_as_float(__builtin_amdgcn_update_dpp(0, __float_as_int(x), 0x4E, 0xF, 0xF, true));
    x += t;
    t = __int_as_float(__builtin_amdgcn_update_dpp(0, __float_as_int(x), 0x141, 0xF, 0xF, true));
    x += t;
    t = __int_as_float(__builtin_amdgcn_update_dpp(0, __float_as_int(x), 0x140, 0xF, 0xF, true));
    return x + t;
}

// (reg r, half h) row-state -> row-per-lane scalar
__device__ __forceinline__ float bcast_row(const float* v, int n, int h) {
    float a = v[0];
#pragma unroll
    for (int r = 1; r < 8; ++r) a = ((n & 7) == r) ? v[r] : a;
    const float ax = __shfl_xor(a, 16, 32);
    return ((n >> 3) == h) ? a : ax;
}

// ---- CDNA5 async global->LDS copy (ASYNCcnt-tracked) ----------------------
__device__ __forceinline__ void async_load_b128(unsigned lds_off, const void* gaddr) {
    asm volatile("global_load_async_to_lds_b128 %0, %1, off"
                 :: "v"(lds_off), "v"(gaddr) : "memory");
}
__device__ __forceinline__ void wait_async_le8() {
    asm volatile("s_wait_asynccnt 8" ::: "memory");
}
__device__ __forceinline__ void wait_async_0() {
    asm volatile("s_wait_asynccnt 0" ::: "memory");
}

// LDS layout (dynamic, base assumed at LDS offset 0; rows padded vs 64 banks)
#define KSTR 136   // 128 + 8 halfword pad  (272B rows -> bank stride 4)
#define VSTR 72    // 64 + 8 halfword pad   (144B rows -> bank stride 36)
#define PSTR 72
struct LdsMap {
    unsigned short k[2][KTILE][KSTR];   // K tile, [key][dim]
    unsigned short v[2][DH][VSTR];      // VT tile, [dim][key]
    unsigned short p[WAVES][16][PSTR];  // per-wave P staging, [row][key]
};

// ---------------------------------------------------------------------------
// Flash attention: 8 waves/block share double-buffered K/VT tiles in LDS.
// Per 64-key tile: 16 WMMAs (S), DPP softmax, P transposed via LDS, 16 WMMAs
// (O^T update).  grid: (NTOK/128, BATCH), block 256, dynLDS = sizeof(LdsMap).
// ---------------------------------------------------------------------------
__launch_bounds__(256)
__global__ void flash_attn_kernel(const unsigned short* __restrict__ Q,
                                  const unsigned short* __restrict__ K,
                                  const unsigned short* __restrict__ VT,
                                  float* __restrict__ O) {
    extern __shared__ char smem[];
    LdsMap* L = (LdsMap*)smem;

    const int lane = threadIdx.x & 31;
    const int wv   = threadIdx.x >> 5;
    const int b    = blockIdx.y;
    const int qbase = (blockIdx.x * WAVES + wv) * 16;
    const int n = lane & 15, h = lane >> 4;

    const unsigned short* Qb  = Q  + (size_t)b * NTOK * DH;
    const unsigned short* Kb  = K  + (size_t)b * NTOK * DH;
    const unsigned short* VTb = VT + (size_t)b * DH * NTOK;
    float* Ob = O + (size_t)b * NTOK * DH;

    // issue one 64-key tile copy: 8 async b128 per wave (K: 2 rows/instr,
    // VT: 4 rows/instr)
    auto issue_tile = [&](int ib, int kt) {
        const int kr0 = wv * 8 + (lane >> 4);
        const int koff = (lane & 15) * 16;
#pragma unroll
        for (int j = 0; j < 4; ++j) {
            const int row = kr0 + 2 * j;
            async_load_b128((unsigned)((char*)&L->k[ib][row][0] - smem) + koff,
                            (const char*)Kb + (size_t)(kt + row) * 256 + koff);
        }
        const int vr0 = wv * 16 + (lane >> 3);
        const int voff = (lane & 7) * 16;
#pragma unroll
        for (int j = 0; j < 4; ++j) {
            const int row = vr0 + 4 * j;
            async_load_b128((unsigned)((char*)&L->v[ib][row][0] - smem) + voff,
                            (const char*)VTb + ((size_t)row * NTOK + kt) * 2 + voff);
        }
    };

    // Q A-fragments (16-bit A 16x32 layout), 4 chunks of 32 dims
    BF16x16 aQ[4];
    {
        const unsigned short* qrow = Qb + (size_t)(qbase + n) * DH;
#pragma unroll
        for (int c = 0; c < 4; ++c) {
            aQ[c].h[0] = *(const us8*)(qrow + 32 * c + 8 * h);
            aQ[c].h[1] = *(const us8*)(qrow + 32 * c + 16 + 8 * h);
        }
    }

    v8f oacc[8];
#pragma unroll
    for (int g = 0; g < 8; ++g)
#pragma unroll
        for (int r = 0; r < 8; ++r) oacc[g][r] = 0.0f;
    float m8[8], l8[8];
#pragma unroll
    for (int r = 0; r < 8; ++r) { m8[r] = -1e30f; l8[r] = 0.0f; }

    const float sc = 0.08838834764831845f * 1.4426950408889634f; // 1/sqrt(128)*log2(e)

    issue_tile(0, 0);

    int ib = 0;
    for (int kt = 0; kt < NTOK; kt += KTILE, ib ^= 1) {
        if (kt + KTILE < NTOK) { issue_tile(ib ^ 1, kt + KTILE); wait_async_le8(); }
        else                   { wait_async_0(); }
        __syncthreads();

        const unsigned short* kb = &L->k[ib][0][0];
        const unsigned short* vb = &L->v[ib][0][0];

        // ---- S = Q * K^T : 4 key sub-tiles x 4 dim chunks -----------------
        v8f s[4];
#pragma unroll
        for (int t = 0; t < 4; ++t)
#pragma unroll
            for (int r = 0; r < 8; ++r) s[t][r] = 0.0f;
#pragma unroll
        for (int c = 0; c < 4; ++c)
#pragma unroll
            for (int t = 0; t < 4; ++t) {
                BF16x16 bk;
                bk.h[0] = *(const us8*)(kb + (16 * t + n) * KSTR + 32 * c + 8 * h);
                bk.h[1] = *(const us8*)(kb + (16 * t + n) * KSTR + 32 * c + 8 * h + 16);
                s[t] = __builtin_amdgcn_wmma_f32_16x16x32_bf16(
                           false, aQ[c].v, false, bk.v, (short)0, s[t], false, false);
            }

        // ---- online softmax, DPP reductions over the 16-lane column dim ---
        float pmax[8];
#pragma unroll
        for (int r = 0; r < 8; ++r)
            pmax[r] = fmaxf(fmaxf(s[0][r], s[1][r]), fmaxf(s[2][r], s[3][r]));
#pragma unroll
        for (int r = 0; r < 8; ++r) pmax[r] = red_max16(pmax[r]);

        float alpha[8];
#pragma unroll
        for (int r = 0; r < 8; ++r) {
            const float mn = fmaxf(m8[r], pmax[r]);
            alpha[r] = __builtin_amdgcn_exp2f((m8[r] - mn) * sc);
            m8[r] = mn;
        }
        float p[4][8], rsum[8];
#pragma unroll
        for (int t = 0; t < 4; ++t)
#pragma unroll
            for (int r = 0; r < 8; ++r)
                p[t][r] = __builtin_amdgcn_exp2f((s[t][r] - m8[r]) * sc);
#pragma unroll
        for (int r = 0; r < 8; ++r)
            rsum[r] = red_sum16((p[0][r] + p[1][r]) + (p[2][r] + p[3][r]));
#pragma unroll
        for (int r = 0; r < 8; ++r) l8[r] = l8[r] * alpha[r] + rsum[r];

        // ---- transpose P through per-wave LDS -----------------------------
        unsigned short* pw = &L->p[wv][0][0];
#pragma unroll
        for (int t = 0; t < 4; ++t)
#pragma unroll
            for (int r = 0; r < 8; ++r)
                pw[(8 * h + r) * PSTR + 16 * t + n] = f2bf(p[t][r]);

        const float a = bcast_row(alpha, n, h);

        BF16x16 bP[2];   // B-frags of P^T, key chunks of 32
#pragma unroll
        for (int kc = 0; kc < 2; ++kc) {
            bP[kc].h[0] = *(const us8*)(pw + n * PSTR + 32 * kc + 8 * h);
            bP[kc].h[1] = *(const us8*)(pw + n * PSTR + 32 * kc + 8 * h + 16);
        }

        // ---- O^T update: 8 dim chunks x 2 key chunks ----------------------
#pragma unroll
        for (int g = 0; g < 8; ++g) {
            v8f cacc = oacc[g];
#pragma unroll
            for (int r = 0; r < 8; ++r) cacc[r] *= a;
#pragma unroll
            for (int kc = 0; kc < 2; ++kc) {
                BF16x16 aV;  // A-frag of V^T chunk [16 dims x 32 keys]
                aV.h[0] = *(const us8*)(vb + (16 * g + n) * VSTR + 32 * kc + 8 * h);
                aV.h[1] = *(const us8*)(vb + (16 * g + n) * VSTR + 32 * kc + 8 * h + 16);
                cacc = __builtin_amdgcn_wmma_f32_16x16x32_bf16(
                           false, aV.v, false, bP[kc].v, (short)0, cacc, false, false);
            }
            oacc[g] = cacc;
        }
        __syncthreads();   // all waves done reading buf ib before it is refilled
    }

    // ---- finalize ---------------------------------------------------------
    const float lv = bcast_row(l8, n, h);
    const float inv = 1.0f / lv;
    float* orow = Ob + (size_t)(qbase + n) * DH;
#pragma unroll
    for (int g = 0; g < 8; ++g) {
        float4 v0 = make_float4(oacc[g][0] * inv, oacc[g][1] * inv,
                                oacc[g][2] * inv, oacc[g][3] * inv);
        float4 v1 = make_float4(oacc[g][4] * inv, oacc[g][5] * inv,
                                oacc[g][6] * inv, oacc[g][7] * inv);
        *(float4*)(orow + 16 * g + 8 * h)     = v0;
        *(float4*)(orow + 16 * g + 8 * h + 4) = v1;
    }
}

// ---------------------------------------------------------------------------
// 1x1-conv projections as fp32 WMMA GEMMs (v_wmma_f32_16x16x4_f32).
// Per wave: one 16-out-channel row tile x 128 pixels (8 accumulators).
// A-frag: 2 consecutive fp32 of W per lane; B-frag: 2 coalesced fp32 of X.
// DUAL: second projection (V) sharing the same X fragments, written to the
// transposed VT layout.  grid: (HW/128, BATCH), block 256.
// ---------------------------------------------------------------------------
template <bool DUAL>
__launch_bounds__(256)
__global__ void gemm_proj_kernel(const float* __restrict__ X,
                                 const float* __restrict__ W1,
                                 const float* __restrict__ B1,
                                 unsigned short* __restrict__ outN,
                                 const float* __restrict__ W2,
                                 const float* __restrict__ B2,
                                 unsigned short* __restrict__ outT,
                                 int cin) {
    const int lane = threadIdx.x & 31;
    const int rt   = threadIdx.x >> 5;      // row tile: out channels 16*rt..+15
    const int b    = blockIdx.y;
    const int g    = blockIdx.x;            // 128-pixel group
    const int n = lane & 15, h = lane >> 4;
    const float* Xb = X + (size_t)b * cin * HW;

    v8f acc1[8], acc2[8];
#pragma unroll
    for (int ps = 0; ps < 8; ++ps)
#pragma unroll
        for (int r = 0; r < 8; ++r) {
            acc1[ps][r] = B1[16 * rt + 8 * h + r];
            if (DUAL) acc2[ps][r] = B2[16 * rt + 8 * h + r];
        }

    for (int c0 = 0; c0 < cin; c0 += 4) {
        const v2f a1 = *(const v2f*)(W1 + (size_t)(16 * rt + n) * cin + c0 + 2 * h);
        v2f a2;
        if (DUAL) a2 = *(const v2f*)(W2 + (size_t)(16 * rt + n) * cin + c0 + 2 * h);
#pragma unroll
        for (int ps = 0; ps < 8; ++ps) {
            const int s = g * 128 + ps * 16 + n;
            v2f bx;
            bx[0] = Xb[(size_t)(c0 + 2 * h) * HW + s];
            bx[1] = Xb[(size_t)(c0 + 2 * h + 1) * HW + s];
            acc1[ps] = __builtin_amdgcn_wmma_f32_16x16x4_f32(
                           false, a1, false, bx, (short)0, acc1[ps], false, false);
            if (DUAL)
                acc2[ps] = __builtin_amdgcn_wmma_f32_16x16x4_f32(
                               false, a2, false, bx, (short)0, acc2[ps], false, false);
        }
    }

    const size_t ob = (size_t)b * DH * HW;
#pragma unroll
    for (int ps = 0; ps < 8; ++ps) {
        const int s = g * 128 + ps * 16 + n;
#pragma unroll
        for (int r = 0; r < 8; ++r) {
            const int c = 16 * rt + 8 * h + r;
            outN[ob + (size_t)c * HW + s] = f2bf(acc1[ps][r]);
            if (DUAL)   // VT[d][ntok], d = s&127 = ps*16+n, ntok = c*32 + g
                outT[ob + (size_t)(ps * 16 + n) * NTOK + c * 32 + g] = f2bf(acc2[ps][r]);
        }
    }
}

// Output projection: out = wo @ O_reshaped + bo + x_lower (fp32 WMMA + fused
// bias/residual).  O element [d,s] = Oflat[d*HW + s].
__launch_bounds__(256)
__global__ void gemm_out_kernel(const float* __restrict__ O,
                                const float* __restrict__ Wo,
                                const float* __restrict__ Bo,
                                const float* __restrict__ resid,
                                float* __restrict__ out) {
    const int lane = threadIdx.x & 31;
    const int rt   = threadIdx.x >> 5;
    const int b    = blockIdx.y;
    const int g    = blockIdx.x;
    const int n = lane & 15, h = lane >> 4;
    const float* Ob = O + (size_t)b * DH * HW;

    v8f acc[8];
#pragma unroll
    for (int ps = 0; ps < 8; ++ps)
#pragma unroll
        for (int r = 0; r < 8; ++r) acc[ps][r] = Bo[16 * rt + 8 * h + r];

    for (int c0 = 0; c0 < DH; c0 += 4) {
        const v2f a = *(const v2f*)(Wo + (size_t)(16 * rt + n) * DH + c0 + 2 * h);
#pragma unroll
        for (int ps = 0; ps < 8; ++ps) {
            const int s = g * 128 + ps * 16 + n;
            v2f bx;
            bx[0] = Ob[(size_t)(c0 + 2 * h) * HW + s];
            bx[1] = Ob[(size_t)(c0 + 2 * h + 1) * HW + s];
            acc[ps] = __builtin_amdgcn_wmma_f32_16x16x4_f32(
                          false, a, false, bx, (short)0, acc[ps], false, false);
        }
    }

    const size_t ob = (size_t)b * DH * HW;
#pragma unroll
    for (int ps = 0; ps < 8; ++ps) {
        const int s = g * 128 + ps * 16 + n;
#pragma unroll
        for (int r = 0; r < 8; ++r) {
            const size_t idx = ob + (size_t)(16 * rt + 8 * h + r) * HW + s;
            out[idx] = acc[ps][r] + resid[idx];
        }
    }
}

// ---------------------------------------------------------------------------
extern "C" void kernel_launch(void* const* d_in, const int* in_sizes, int n_in,
                              void* d_out, int out_size, void* d_ws, size_t ws_size,
                              hipStream_t stream) {
    const float* x_upper = (const float*)d_in[0];
    const float* x_lower = (const float*)d_in[1];
    const float* wq = (const float*)d_in[2];
    const float* bq = (const float*)d_in[3];
    const float* wk = (const float*)d_in[4];
    const float* bk = (const float*)d_in[5];
    const float* wv = (const float*)d_in[6];
    const float* bv = (const float*)d_in[7];
    const float* wo = (const float*)d_in[8];
    const float* bo = (const float*)d_in[9];
    float* out = (float*)d_out;

    // workspace: Q(4MB bf16) | K(4MB bf16) | VT(4MB bf16) | O(8MB fp32)
    char* ws = (char*)d_ws;
    unsigned short* Q  = (unsigned short*)(ws);
    unsigned short* K  = (unsigned short*)(ws + (size_t)4  * 1024 * 1024);
    unsigned short* VT = (unsigned short*)(ws + (size_t)8  * 1024 * 1024);
    float*          O  = (float*)         (ws + (size_t)12 * 1024 * 1024);

    const dim3 blk(256);
    const dim3 gg(HW / 128, BATCH);

    gemm_proj_kernel<false><<<gg, blk, 0, stream>>>(
        x_upper, wq, bq, Q, nullptr, nullptr, nullptr, CIN);
    gemm_proj_kernel<true><<<gg, blk, 0, stream>>>(
        x_lower, wk, bk, K, wv, bv, VT, 2 * CIN);

    const dim3 gattn(NTOK / (16 * WAVES), BATCH);
    flash_attn_kernel<<<gattn, dim3(256), sizeof(LdsMap), stream>>>(Q, K, VT, O);

    gemm_out_kernel<<<gg, blk, 0, stream>>>(O, wo, bo, x_lower, out);
}
